// CausalSelfAttention_64089501990917
// MI455X (gfx1250) — matrix operand order repaired and checked by reading, not compiled, not run on hardware
//
#include <hip/hip_runtime.h>
#include <hip/hip_bf16.h>
#include <math.h>

#define EMBED  2048
#define NHEADS 16
#define HDIM   128
#define BSZ    2
#define TLEN   2048
#define MROWS  (BSZ * TLEN)   // 4096

typedef __attribute__((ext_vector_type(16))) __bf16 v16bf;
typedef __attribute__((ext_vector_type(8)))  float  v8f;

struct AFrag { union { v16bf v; unsigned u[8]; }; };

__device__ __forceinline__ v8f wmma_bf16(const AFrag& a, const AFrag& b, v8f c) {
  return __builtin_amdgcn_wmma_f32_16x16x32_bf16(false, a.v, false, b.v, (short)0, c,
                                                 false, false);
}

// A-matrix 16x32 bf16 fragment (ISA 7.12.2): lane L holds row M=L%16.
// lanes 0-15: V0..V3 -> K=(0,1)..(6,7); V4..V7 -> K=(16,17)..(22,23)
// lanes16-31: V0..V3 -> K=(8,9)..(14,15); V4..V7 -> K=(24,25)..(30,31)
__device__ __forceinline__ AFrag load_a_frag(const __bf16* __restrict__ base, int ld,
                                             int row0, int k0, int lane) {
  AFrag f;
  const int m  = row0 + (lane & 15);
  const int kh = (lane >> 4) << 3;  // 0 or 8
  const unsigned* p0 = (const unsigned*)(base + (size_t)m * ld + k0 + kh);
  const unsigned* p1 = (const unsigned*)(base + (size_t)m * ld + k0 + 16 + kh);
#pragma unroll
  for (int v = 0; v < 4; ++v) { f.u[v] = p0[v]; f.u[4 + v] = p1[v]; }
  return f;
}

// B-matrix 32x16 bf16 fragment: lane n holds column N=n%16.
// lanes 0-15: V0..V3 -> K=0..7, V4..V7 -> K=16..23; lanes 16-31: K=8..15, K=24..31.
// Source row-major Bstore[n][k] == B[k][n] (torch Linear W[out][in]).
__device__ __forceinline__ AFrag load_b_frag(const __bf16* __restrict__ base, int ld,
                                             int n0, int k0, int lane) {
  AFrag f;
  const int n  = n0 + (lane & 15);
  const int kh = (lane >> 4) << 3;  // 0 or 8
  const unsigned* p0 = (const unsigned*)(base + (size_t)n * ld + k0 + kh);
  const unsigned* p1 = (const unsigned*)(base + (size_t)n * ld + k0 + 16 + kh);
#pragma unroll
  for (int v = 0; v < 4; ++v) { f.u[v] = p0[v]; f.u[4 + v] = p1[v]; }
  return f;
}

// Same fragment shape, but tile-local from a [64][32] LDS tile.
__device__ __forceinline__ AFrag load_b_frag_lds(const __bf16* tile, int n0, int lane) {
  AFrag f;
  const int n  = n0 + (lane & 15);
  const int kh = (lane >> 4) << 3;  // 0 or 8
  const unsigned* p0 = (const unsigned*)(tile + n * 32 + kh);
  const unsigned* p1 = (const unsigned*)(tile + n * 32 + 16 + kh);
#pragma unroll
  for (int v = 0; v < 4; ++v) { f.u[v] = p0[v]; f.u[4 + v] = p1[v]; }
  return f;
}

__global__ void f32_to_bf16_kernel(const float* __restrict__ in,
                                   __bf16* __restrict__ out, int n) {
  for (int i = blockIdx.x * blockDim.x + threadIdx.x; i < n;
       i += gridDim.x * blockDim.x)
    out[i] = (__bf16)in[i];
}

// C = A[MROWS x EMBED] @ W[EMBED x EMBED]^T, W stored [out,in].
// 4 waves/block: each wave owns 32 rows; the 64-col x 32-k W tile is shared via
// LDS, staged with double-buffered gfx1250 async loads (ASYNCcnt).
// OUT_MODE 0: f32 row-major [MROWS,EMBED]
// OUT_MODE 1: bf16 [B,H,T,D]
// OUT_MODE 2: bf16 [B,H,D,T]   (transposed V for the PV matmul)
template <int OUT_MODE>
__global__ __launch_bounds__(128) void gemm_bf16_kernel(const __bf16* __restrict__ A,
                                                        const __bf16* __restrict__ W,
                                                        void* __restrict__ out) {
  const int lane = threadIdx.x & 31;
  const int wave = threadIdx.x >> 5;
  const int row0 = (blockIdx.x >> 5) * 128 + wave * 32;  // 32 row-blocks of 128
  const int col0 = (blockIdx.x & 31) * 64;               // 32 col-blocks of 64

  __shared__ __bf16 Btile[2][64][32];  // 8 KB double-buffered W tile

  // Each of the 128 threads async-copies two 16B chunks of the 4KB tile.
  auto issue = [&](int buf, int k0) {
#pragma unroll
    for (int i = 0; i < 2; ++i) {
      const int c  = threadIdx.x + i * 128;   // 0..255 chunks
      const int n  = c >> 2;                  // row in tile (0..63)
      const int kc = (c & 3) * 8;             // bf16 offset in row (16B chunks)
      const unsigned lds_off = (unsigned)(size_t)&Btile[buf][n][kc];
      const unsigned long long g =
          (unsigned long long)(size_t)(W + (size_t)(col0 + n) * EMBED + k0 + kc);
      asm volatile("global_load_async_to_lds_b128 %0, %1, off"
                   :: "v"(lds_off), "v"(g) : "memory");
    }
  };

  v8f acc[2][4];
#pragma unroll
  for (int i = 0; i < 2; ++i)
#pragma unroll
    for (int j = 0; j < 4; ++j)
      acc[i][j] = (v8f){0.f, 0.f, 0.f, 0.f, 0.f, 0.f, 0.f, 0.f};

  issue(0, 0);
  for (int k0 = 0; k0 < EMBED; k0 += 32) {
    const int cur = (k0 >> 5) & 1;
    __syncthreads();  // all waves done reading buffer cur^1 (prev iteration)
    if (k0 + 32 < EMBED) {
      issue(cur ^ 1, k0 + 32);
      asm volatile("s_wait_asynccnt 0x2" ::: "memory");  // cur's copies landed
    } else {
      asm volatile("s_wait_asynccnt 0x0" ::: "memory");
    }
    __syncthreads();  // every wave's share of buffer cur is in LDS

    if (k0 + 32 < EMBED)  // prefetch next A slice (gfx1250 global_prefetch)
      __builtin_prefetch(A + (size_t)(row0 + lane) * EMBED + k0 + 32, 0, 0);

    AFrag a0 = load_a_frag(A, EMBED, row0, k0, lane);
    AFrag a1 = load_a_frag(A, EMBED, row0 + 16, k0, lane);
    const __bf16* bt = &Btile[cur][0][0];
    AFrag b0 = load_b_frag_lds(bt,  0, lane);
    AFrag b1 = load_b_frag_lds(bt, 16, lane);
    AFrag b2 = load_b_frag_lds(bt, 32, lane);
    AFrag b3 = load_b_frag_lds(bt, 48, lane);
    acc[0][0] = wmma_bf16(a0, b0, acc[0][0]);
    acc[0][1] = wmma_bf16(a0, b1, acc[0][1]);
    acc[0][2] = wmma_bf16(a0, b2, acc[0][2]);
    acc[0][3] = wmma_bf16(a0, b3, acc[0][3]);
    acc[1][0] = wmma_bf16(a1, b0, acc[1][0]);
    acc[1][1] = wmma_bf16(a1, b1, acc[1][1]);
    acc[1][2] = wmma_bf16(a1, b2, acc[1][2]);
    acc[1][3] = wmma_bf16(a1, b3, acc[1][3]);
  }

  // C/D layout: VGPR r, lanes0-15 -> M=r, lanes16-31 -> M=r+8; N = lane%16.
#pragma unroll
  for (int i = 0; i < 2; ++i)
#pragma unroll
    for (int j = 0; j < 4; ++j)
#pragma unroll
      for (int r = 0; r < 8; ++r) {
        int row = row0 + 16 * i + r + ((lane >> 4) << 3);
        int col = col0 + 16 * j + (lane & 15);
        float val = acc[i][j][r];
        if (OUT_MODE == 0) {
          ((float*)out)[(size_t)row * EMBED + col] = val;
        } else {
          int b = row >> 11, t = row & (TLEN - 1);
          int h = col >> 7,  d = col & (HDIM - 1);
          size_t idx = (OUT_MODE == 1)
              ? ((((size_t)b * NHEADS + h) * TLEN + t) * HDIM + d)
              : ((((size_t)b * NHEADS + h) * HDIM + d) * TLEN + t);
          ((__bf16*)out)[idx] = (__bf16)val;
        }
      }
}

// Flash attention: one wave per (b,h, 16-query tile). Online softmax in f32.
__global__ __launch_bounds__(32) void attn_kernel(const __bf16* __restrict__ Q,
                                                  const __bf16* __restrict__ Kc,
                                                  const __bf16* __restrict__ Vt,
                                                  __bf16* __restrict__ O) {
  const int lane  = threadIdx.x;
  const int bh    = blockIdx.x >> 7;     // 32 (b,h) pairs
  const int qbase = (blockIdx.x & 127) * 16;
  const int b = bh >> 4, h = bh & 15;
  const __bf16* Qp = Q  + (size_t)bh * TLEN * HDIM;   // [t][d]
  const __bf16* Kp = Kc + (size_t)bh * TLEN * HDIM;   // [t][d]
  const __bf16* Vp = Vt + (size_t)bh * HDIM * TLEN;   // [d][t]

  __shared__ __bf16 Plds[16][32];

  AFrag qf[4];
#pragma unroll
  for (int c = 0; c < 4; ++c) qf[c] = load_a_frag(Qp, HDIM, qbase, c * 32, lane);

  v8f o[8];
#pragma unroll
  for (int dc = 0; dc < 8; ++dc) o[dc] = (v8f){0.f,0.f,0.f,0.f,0.f,0.f,0.f,0.f};
  float m[8], l[8];
#pragma unroll
  for (int r = 0; r < 8; ++r) { m[r] = -INFINITY; l[r] = 0.f; }

  const float scale = 0.08838834764831845f;  // 1/sqrt(128)
  const int ncol = lane & 15;
  const int half = (lane >> 4) << 3;         // 0 or 8 (row offset of this lane half)

  for (int j0 = 0; j0 <= qbase + 15; j0 += 32) {
    // S = Q K^T over d (K-dim), two 16x16 N-tiles covering 32 keys
    v8f s0 = (v8f){0.f,0.f,0.f,0.f,0.f,0.f,0.f,0.f};
    v8f s1 = s0;
#pragma unroll
    for (int c = 0; c < 4; ++c) {
      AFrag kb0 = load_b_frag(Kp, HDIM, j0,      c * 32, lane);
      AFrag kb1 = load_b_frag(Kp, HDIM, j0 + 16, c * 32, lane);
      s0 = wmma_bf16(qf[c], kb0, s0);
      s1 = wmma_bf16(qf[c], kb1, s1);
    }

    const bool need_mask = (j0 + 31) > qbase;
    float alpha[8];
#pragma unroll
    for (int r = 0; r < 8; ++r) {
      const int qr = qbase + r + half;
      float a  = s0[r] * scale;
      float bb = s1[r] * scale;
      if (need_mask) {
        if (j0 + ncol      > qr) a  = -INFINITY;
        if (j0 + 16 + ncol > qr) bb = -INFINITY;
      }
      // row max across the 16 lanes of this half (xor masks < 16 stay in-half)
      float t = fmaxf(a, bb);
      t = fmaxf(t, __shfl_xor(t, 1));
      t = fmaxf(t, __shfl_xor(t, 2));
      t = fmaxf(t, __shfl_xor(t, 4));
      t = fmaxf(t, __shfl_xor(t, 8));
      const float mn = fmaxf(m[r], t);
      alpha[r] = __expf(m[r] - mn);
      const float p0 = __expf(a - mn);
      const float p1 = __expf(bb - mn);
      float rs = p0 + p1;
      rs += __shfl_xor(rs, 1);
      rs += __shfl_xor(rs, 2);
      rs += __shfl_xor(rs, 4);
      rs += __shfl_xor(rs, 8);
      l[r] = l[r] * alpha[r] + rs;
      m[r] = mn;
      s0[r] = p0;
      s1[r] = p1;
    }
#pragma unroll
    for (int dc = 0; dc < 8; ++dc)
#pragma unroll
      for (int r = 0; r < 8; ++r) o[dc][r] *= alpha[r];

    // Stage P (C-layout) -> LDS -> A-fragment layout
#pragma unroll
    for (int r = 0; r < 8; ++r) {
      const int row = r + half;
      Plds[row][ncol]      = (__bf16)s0[r];
      Plds[row][16 + ncol] = (__bf16)s1[r];
    }
    __syncthreads();
    AFrag pf;
    {
      const int mrow = lane & 15;
      const int kh   = (lane >> 4) << 3;
      const unsigned* pl0 = (const unsigned*)&Plds[mrow][kh];
      const unsigned* pl1 = (const unsigned*)&Plds[mrow][16 + kh];
#pragma unroll
      for (int v = 0; v < 4; ++v) { pf.u[v] = pl0[v]; pf.u[4 + v] = pl1[v]; }
    }
    __syncthreads();

    // O += P @ V : K-dim = 32 keys, N-dim = 16 of 128 head dims per WMMA
#pragma unroll
    for (int dc = 0; dc < 8; ++dc) {
      AFrag vb = load_b_frag(Vp, TLEN, dc * 16, j0, lane);
      o[dc] = wmma_bf16(pf, vb, o[dc]);
    }
  }

  // normalize + store concat-head layout [b*T+t][h*128+d] (bf16 for final GEMM)
  float linv[8];
#pragma unroll
  for (int r = 0; r < 8; ++r) linv[r] = 1.0f / l[r];
#pragma unroll
  for (int dc = 0; dc < 8; ++dc)
#pragma unroll
    for (int r = 0; r < 8; ++r) {
      const int qr  = qbase + r + half;
      const int col = h * HDIM + dc * 16 + ncol;
      O[((size_t)b * TLEN + qr) * EMBED + col] = (__bf16)(o[dc][r] * linv[r]);
    }
}

extern "C" void kernel_launch(void* const* d_in, const int* in_sizes, int n_in,
                              void* d_out, int out_size, void* d_ws, size_t ws_size,
                              hipStream_t stream) {
  const float* x  = (const float*)d_in[0];
  const float* Wq = (const float*)d_in[1];
  const float* Wk = (const float*)d_in[2];
  const float* Wv = (const float*)d_in[3];
  const float* Wo = (const float*)d_in[4];

  char* ws = (char*)d_ws;
  size_t off = 0;
  const size_t XB = (size_t)MROWS * EMBED * sizeof(__bf16);   // 16.8 MB
  const size_t WB = (size_t)EMBED * EMBED * sizeof(__bf16);   //  8.4 MB
  __bf16* xb  = (__bf16*)(ws + off); off += XB;
  __bf16* Wqb = (__bf16*)(ws + off); off += WB;
  __bf16* Wkb = (__bf16*)(ws + off); off += WB;
  __bf16* Wvb = (__bf16*)(ws + off); off += WB;
  __bf16* Wob = (__bf16*)(ws + off); off += WB;
  __bf16* Qb  = (__bf16*)(ws + off); off += XB;
  __bf16* Kb  = (__bf16*)(ws + off); off += XB;
  __bf16* Vt  = (__bf16*)(ws + off); off += XB;
  __bf16* Ob  = (__bf16*)(ws + off); off += XB;

  f32_to_bf16_kernel<<<2048, 256, 0, stream>>>(x,  xb,  MROWS * EMBED);
  f32_to_bf16_kernel<<<2048, 256, 0, stream>>>(Wq, Wqb, EMBED * EMBED);
  f32_to_bf16_kernel<<<2048, 256, 0, stream>>>(Wk, Wkb, EMBED * EMBED);
  f32_to_bf16_kernel<<<2048, 256, 0, stream>>>(Wv, Wvb, EMBED * EMBED);
  f32_to_bf16_kernel<<<2048, 256, 0, stream>>>(Wo, Wob, EMBED * EMBED);

  // 1024 blocks of 4 waves; each block: 128 rows x 64 cols
  gemm_bf16_kernel<1><<<1024, 128, 0, stream>>>(xb, Wqb, (void*)Qb);  // Q [b,h,t,d]
  gemm_bf16_kernel<1><<<1024, 128, 0, stream>>>(xb, Wkb, (void*)Kb);  // K [b,h,t,d]
  gemm_bf16_kernel<2><<<1024, 128, 0, stream>>>(xb, Wvb, (void*)Vt);  // V [b,h,d,t]

  attn_kernel<<<BSZ * NHEADS * (TLEN / 16), 32, 0, stream>>>(Qb, Kb, Vt, Ob);

  gemm_bf16_kernel<0><<<1024, 128, 0, stream>>>(Ob, Wob, d_out);      // f32 out
}
